// ECAPA_RetNet_4114578670501
// MI455X (gfx1250) — matrix operand order repaired and checked by reading, not compiled, not run on hardware
//
#include <hip/hip_runtime.h>
#include <math.h>

// ---------------------------------------------------------------------------
// ECAPA-RetNet forward on gfx1250 (MI455X): bf16 WMMA everywhere, flash-style
// causal retention (no S x S materialization), fp32 accumulation.
// Async-to-LDS staging (ASYNCcnt) for GEMM A tiles.
// ---------------------------------------------------------------------------

typedef __bf16 bf16;
typedef __attribute__((ext_vector_type(16))) __bf16 bf16x16;
typedef __attribute__((ext_vector_type(8)))  float  f32x8;

struct U8 { unsigned u[8]; };
static_assert(sizeof(U8) == sizeof(bf16x16), "frag size");

#define LAYERS 4
#define DMODEL 192
#define FDIM   768
#define HEADS  8
#define HDIM   24
#define HPAD   32
#define OUTD   1024
#define SEQ    1024
#define BATCH  8
#define ROWS   (BATCH * SEQ)   // 8192 tokens
#define EPS    1e-5f

__device__ __forceinline__ f32x8 wmma_bf16(const U8& a, const U8& b, f32x8 c) {
  return __builtin_amdgcn_wmma_f32_16x16x32_bf16(
      false, __builtin_bit_cast(bf16x16, a),
      false, __builtin_bit_cast(bf16x16, b),
      (short)0, c, false, false);
}

// A-matrix (16x32 bf16) per-lane K index of dword-pair p (contiguous: k = 16*hf + 2p, matches B frags)
__device__ __forceinline__ int a_kpair(int p, int hf) {
  return hf * 16 + 2 * p;
}

// async copy 16B from global to LDS (tracked on ASYNCcnt)
__device__ __forceinline__ void async_b128(unsigned lds_off, const void* gptr) {
  asm volatile("global_load_async_to_lds_b128 %0, %1, off"
               :: "v"(lds_off), "v"(gptr) : "memory");
}
__device__ __forceinline__ void wait_async0() {
  asm volatile("s_wait_asynccnt 0x0" ::: "memory");
}

// ---------------------------------------------------------------------------
// Generic bf16 GEMM:  C[M,N] (f32) = A[M,K](bf16) * B[K,N](bf16) (+ bias)
// block = 128 threads (4 waves), tile 64x64, K-step 32.
// A tile staged via global_load_async_to_lds_b128; B tile transposed to
// N-major LDS through VGPRs (2-byte transpose granularity).
// ---------------------------------------------------------------------------
__global__ __launch_bounds__(128) void gemm_bf16_kernel(
    const bf16* __restrict__ A, const bf16* __restrict__ Bm,
    float* __restrict__ C, int M, int N, int K, const float* __restrict__ bias)
{
  __shared__ __align__(16) bf16 As[64][32];   // [m][k]
  __shared__ __align__(16) bf16 Bs[64][32];   // N-major: [n][k]

  const int tid  = threadIdx.x;
  const int wave = tid >> 5;
  const int lane = tid & 31;
  const int hf   = lane >> 4;
  const int l16  = lane & 15;
  const int m0   = blockIdx.y * 64;
  const int n0   = blockIdx.x * 64;
  (void)M;

  f32x8 acc[4] = {};

  for (int k0 = 0; k0 < K; k0 += 32) {
    // async stage A 64x32 bf16 = 4 KB = 256 x 16B chunks, 2 per thread
#pragma unroll
    for (int c = 0; c < 2; ++c) {
      int idx = tid + c * 128;          // 0..255
      int r   = idx >> 2;               // row 0..63
      int col = (idx & 3) * 8;          // bf16 col 0/8/16/24
      async_b128((unsigned)(size_t)&As[r][col],
                 A + (size_t)(m0 + r) * K + k0 + col);
    }
    // stage B 32x64 row-major -> N-major LDS (coalesced over n)
    for (int i = tid; i < 64 * 32; i += 128) {
      int kk = i >> 6, n = i & 63;
      Bs[n][kk] = Bm[(size_t)(k0 + kk) * N + n0 + n];
    }
    if (k0 + 32 < K)   // prefetch next weight tile into L2
      __builtin_prefetch(Bm + (size_t)(k0 + 32) * N + n0 + (tid & 63), 0, 0);
    wait_async0();
    __syncthreads();

    U8 a;
    const bf16* arow = &As[wave * 16 + l16][0];
#pragma unroll
    for (int p = 0; p < 8; ++p)
      a.u[p] = *(const unsigned*)(arow + a_kpair(p, hf));

#pragma unroll
    for (int ns = 0; ns < 4; ++ns) {
      U8 b;
      const bf16* brow = &Bs[ns * 16 + l16][0];
#pragma unroll
      for (int p = 0; p < 8; ++p)
        b.u[p] = *(const unsigned*)(brow + hf * 16 + 2 * p);
      acc[ns] = wmma_bf16(a, b, acc[ns]);
    }
    __syncthreads();
  }

  const int wm = m0 + wave * 16;
#pragma unroll
  for (int ns = 0; ns < 4; ++ns) {
#pragma unroll
    for (int r = 0; r < 8; ++r) {
      int row = wm + r + 8 * hf;
      int col = n0 + ns * 16 + l16;
      float v = acc[ns][r];
      if (bias) v += bias[col];
      C[(size_t)row * N + col] = v;
    }
  }
}

// ---------------------------------------------------------------------------
// Flash-style causal retention for one (batch, head, 64-query-row) strip.
// Q/K bf16 [B][H][S][32]; V bf16 pre-transposed [B][H][32][S] so B-fragments
// read contiguous key-pairs straight from global. Barrier-free inner loop;
// per-wave causal trip count. Y written fp32 as [B*S][192].
// ---------------------------------------------------------------------------
__global__ __launch_bounds__(128) void retention_kernel(
    const bf16* __restrict__ Qb, const bf16* __restrict__ Kb,
    const bf16* __restrict__ Vb, float* __restrict__ Y)
{
  __shared__ __align__(16) bf16 Sb[4][16][32];  // per-wave decayed scores

  const int nt = blockIdx.x;         // query 64-row tile
  const int h  = blockIdx.y;
  const int bb = blockIdx.z;
  const int tid  = threadIdx.x;
  const int wave = tid >> 5;
  const int lane = tid & 31;
  const int hf   = lane >> 4;
  const int l16  = lane & 15;

  const size_t headoff = ((size_t)(bb * HEADS + h)) * SEQ * HPAD;
  const bf16* Qh  = Qb + headoff;
  const bf16* Kh  = Kb + headoff;
  const bf16* VtG = Vb + headoff;    // same element count, [32][S] layout

  const int qr0 = nt * 64 + wave * 16;

  // RetNet decay: gamma_h = 1 - exp(linspace(log 1/32, log 1/512, H))
  const float lg0 = logf(1.0f / 32.0f), lg1 = logf(1.0f / 512.0f);
  const float gamma = 1.0f - expf(lg0 + (float)h * (lg1 - lg0) / (float)(HEADS - 1));
  const float lgam  = logf(gamma);

  // Q fragment (persistent for the whole strip)
  U8 qa;
  {
    const bf16* qrow = Qh + (size_t)(qr0 + l16) * HPAD;
#pragma unroll
    for (int p = 0; p < 8; ++p)
      qa.u[p] = *(const unsigned*)(qrow + a_kpair(p, hf));
  }

  f32x8 y0 = {}, y1 = {};

  for (int mb = 0; mb <= qr0 + 15; mb += 32) {   // per-wave causal bound
    // scores for two 16-key subtiles -> decay -> bf16 into per-wave LDS
#pragma unroll
    for (int ms = 0; ms < 2; ++ms) {
      const int mstart = mb + ms * 16;
      U8 kf;
      const bf16* krow = Kh + (size_t)(mstart + l16) * HPAD;
#pragma unroll
      for (int p = 0; p < 8; ++p)
        kf.u[p] = *(const unsigned*)(krow + hf * 16 + 2 * p);
      f32x8 s = {};
      s = wmma_bf16(qa, kf, s);      // s = Q . K^T  (16x16)
#pragma unroll
      for (int r = 0; r < 8; ++r) {
        int qrow_i = qr0 + r + 8 * hf;
        int kcol   = mstart + l16;
        float v = (qrow_i >= kcol)
                    ? s[r] * __expf((float)(qrow_i - kcol) * lgam)
                    : 0.0f;
        Sb[wave][r + 8 * hf][ms * 16 + l16] = (bf16)v;
      }
    }

    // A-frag from decayed scores (wave-local LDS round trip re-layouts C->A)
    U8 sa;
    const bf16* srow = &Sb[wave][l16][0];
#pragma unroll
    for (int p = 0; p < 8; ++p)
      sa.u[p] = *(const unsigned*)(srow + a_kpair(p, hf));

    // B-frags for V straight from transposed global layout
    U8 v0, v1;
#pragma unroll
    for (int p = 0; p < 8; ++p) {
      int kk = mb + hf * 16 + 2 * p;
      v0.u[p] = *(const unsigned*)(VtG + (size_t)l16 * SEQ + kk);
      v1.u[p] = *(const unsigned*)(VtG + (size_t)(16 + l16) * SEQ + kk);
    }
    y0 = wmma_bf16(sa, v0, y0);      // hd 0..15
    y1 = wmma_bf16(sa, v1, y1);      // hd 16..31 (24..31 zero pad)
  }

  // store Y -> fp32 [b*s][192]
#pragma unroll
  for (int r = 0; r < 8; ++r) {
    int row = qr0 + r + 8 * hf;
    float* yp = Y + ((size_t)(bb * SEQ + row)) * DMODEL + h * HDIM;
    yp[l16] = y0[r];
    if (l16 < 8) yp[16 + l16] = y1[r];
  }
}

// ---------------------------------------------------------------------------
// LayerNorm (wave per row, D=192 -> 6 elems/lane), bf16 output for GEMM A.
// ---------------------------------------------------------------------------
__global__ __launch_bounds__(256) void layernorm_bf16_kernel(
    const float* __restrict__ X, const float* __restrict__ g,
    const float* __restrict__ b, bf16* __restrict__ out)
{
  const int wave = threadIdx.x >> 5;
  const int lane = threadIdx.x & 31;
  const int row  = blockIdx.x * 8 + wave;
  const float* x = X + (size_t)row * DMODEL;

  float vals[6];
  float s = 0.f;
#pragma unroll
  for (int j = 0; j < 6; ++j) { vals[j] = x[lane + 32 * j]; s += vals[j]; }
#pragma unroll
  for (int off = 16; off; off >>= 1) s += __shfl_xor(s, off);
  const float mu = s / (float)DMODEL;

  float vs = 0.f;
#pragma unroll
  for (int j = 0; j < 6; ++j) { float d = vals[j] - mu; vs += d * d; }
#pragma unroll
  for (int off = 16; off; off >>= 1) vs += __shfl_xor(vs, off);
  const float rstd = rsqrtf(vs / (float)DMODEL + EPS);

  bf16* o = out + (size_t)row * DMODEL;
#pragma unroll
  for (int j = 0; j < 6; ++j) {
    int c = lane + 32 * j;
    o[c] = (bf16)((vals[j] - mu) * rstd * g[c] + b[c]);
  }
}

// ---------------------------------------------------------------------------
// Small elementwise / repack kernels
// ---------------------------------------------------------------------------
__global__ void f32_to_bf16_kernel(const float* __restrict__ in,
                                   bf16* __restrict__ out, int n) {
  int i = blockIdx.x * blockDim.x + threadIdx.x;
  if (i < n) out[i] = (bf16)in[i];
}

__global__ void add_inplace_kernel(float* __restrict__ dst,
                                   const float* __restrict__ src, int n) {
  int i = blockIdx.x * blockDim.x + threadIdx.x;
  if (i < n) dst[i] += src[i];
}

__global__ void gelu_bf16_kernel(const float* __restrict__ in,
                                 bf16* __restrict__ out, int n) {
  int i = blockIdx.x * blockDim.x + threadIdx.x;
  if (i < n) {
    float x = in[i];
    float t = tanhf(0.7978845608028654f * (x + 0.044715f * x * x * x));
    out[i] = (bf16)(0.5f * x * (1.0f + t));
  }
}

// pack [Wq|Wk|Wv|Wg] -> bf16 [L][D][4D] for one fused QKVG GEMM
__global__ void pack_qkvg_kernel(const float* __restrict__ Wq,
                                 const float* __restrict__ Wk,
                                 const float* __restrict__ Wv,
                                 const float* __restrict__ Wg,
                                 bf16* __restrict__ out) {
  int i = blockIdx.x * blockDim.x + threadIdx.x;
  const int total = LAYERS * DMODEL * 4 * DMODEL;
  if (i >= total) return;
  int j = i % (4 * DMODEL);
  int k = (i / (4 * DMODEL)) % DMODEL;
  int l = i / (4 * DMODEL * DMODEL);
  int sel = j / DMODEL, c = j % DMODEL;
  const float* W = (sel == 0) ? Wq : (sel == 1) ? Wk : (sel == 2) ? Wv : Wg;
  out[i] = (bf16)W[((size_t)l * DMODEL + k) * DMODEL + c];
}

// QKVG fp32 [8192][768] -> bf16 Q/K [B][H][S][32], V transposed [B][H][32][S]
__global__ void repack_qkv_kernel(const float* __restrict__ QKVG,
                                  bf16* __restrict__ Q, bf16* __restrict__ K,
                                  bf16* __restrict__ V) {
  int i = blockIdx.x * blockDim.x + threadIdx.x;
  const int total = ROWS * HEADS * HPAD;
  if (i >= total) return;
  int d  = i & 31;
  int t  = i >> 5;
  int hh = t & (HEADS - 1);
  int row = t >> 3;                          // b*S + s
  const float* src = QKVG + (size_t)row * (4 * DMODEL);
  float q = 0.f, k = 0.f, v = 0.f;
  if (d < HDIM) {
    q = src[hh * HDIM + d];
    k = src[DMODEL + hh * HDIM + d];
    v = src[2 * DMODEL + hh * HDIM + d];
  }
  int bidx = row >> 10, s = row & (SEQ - 1);
  size_t base = (size_t)(bidx * HEADS + hh);
  size_t o  = (base * SEQ + s) * HPAD + d;        // [B][H][S][32]
  size_t oV = (base * HPAD + d) * SEQ + s;        // [B][H][32][S]
  Q[o] = (bf16)q; K[o] = (bf16)k; V[oV] = (bf16)v;
}

// per-(row, head) group norm over 24 features, times silu(gate), bf16 out
__global__ void gn_gate_kernel(const float* __restrict__ Yr,
                               const float* __restrict__ QKVG,
                               const float* __restrict__ gw,
                               const float* __restrict__ gb,
                               bf16* __restrict__ out) {
  int i = blockIdx.x * blockDim.x + threadIdx.x;
  if (i >= ROWS * HEADS) return;
  int h = i & (HEADS - 1);
  int row = i >> 3;
  const float* y = Yr + (size_t)row * DMODEL + h * HDIM;
  float mu = 0.f;
#pragma unroll
  for (int j = 0; j < HDIM; ++j) mu += y[j];
  mu /= (float)HDIM;
  float var = 0.f;
#pragma unroll
  for (int j = 0; j < HDIM; ++j) { float d = y[j] - mu; var += d * d; }
  float rstd = rsqrtf(var / (float)HDIM + EPS);
  const float* g = QKVG + (size_t)row * (4 * DMODEL) + 3 * DMODEL + h * HDIM;
  bf16* o = out + (size_t)row * DMODEL + h * HDIM;
#pragma unroll
  for (int j = 0; j < HDIM; ++j) {
    float yn = (y[j] - mu) * rstd * gw[h * HDIM + j] + gb[h * HDIM + j];
    float gv = g[j];
    float si = gv / (1.0f + expf(-gv));
    o[j] = (bf16)(si * yn);
  }
}

// ---------------------------------------------------------------------------
// Launcher
// ---------------------------------------------------------------------------
extern "C" void kernel_launch(void* const* d_in, const int* in_sizes, int n_in,
                              void* d_out, int out_size, void* d_ws, size_t ws_size,
                              hipStream_t stream) {
  (void)in_sizes; (void)n_in; (void)out_size; (void)ws_size;
  const float* x    = (const float*)d_in[0];
  const float* ln_g = (const float*)d_in[1];
  const float* ln_b = (const float*)d_in[2];
  const float* Wq   = (const float*)d_in[3];
  const float* Wk   = (const float*)d_in[4];
  const float* Wv   = (const float*)d_in[5];
  const float* Wg   = (const float*)d_in[6];
  const float* Wo   = (const float*)d_in[7];
  const float* gn_w = (const float*)d_in[8];
  const float* gn_b = (const float*)d_in[9];
  const float* W1   = (const float*)d_in[10];
  const float* W2   = (const float*)d_in[11];
  const float* Wenc = (const float*)d_in[12];
  const float* benc = (const float*)d_in[13];

  char* p = (char*)d_ws;
  auto take = [&](size_t bytes) -> char* {
    char* r = p;
    p += (bytes + 255) & ~(size_t)255;
    return r;
  };

  bf16*  wqkvg = (bf16*)take((size_t)LAYERS * DMODEL * 4 * DMODEL * 2);
  bf16*  wob   = (bf16*)take((size_t)LAYERS * DMODEL * DMODEL * 2);
  bf16*  w1b   = (bf16*)take((size_t)LAYERS * DMODEL * FDIM * 2);
  bf16*  w2b   = (bf16*)take((size_t)LAYERS * FDIM * DMODEL * 2);
  bf16*  wencb = (bf16*)take((size_t)DMODEL * OUTD * 2);
  float* XF    = (float*)take((size_t)ROWS * DMODEL * 4);
  float* YF    = (float*)take((size_t)ROWS * DMODEL * 4);
  float* YR    = (float*)take((size_t)ROWS * DMODEL * 4);
  float* CF    = (float*)take((size_t)ROWS * FDIM * 4);
  bf16*  ABuf  = (bf16*)take((size_t)ROWS * FDIM * 2);
  bf16*  Qb    = (bf16*)take((size_t)ROWS * HEADS * HPAD * 2);
  bf16*  Kb    = (bf16*)take((size_t)ROWS * HEADS * HPAD * 2);
  bf16*  Vb    = (bf16*)take((size_t)ROWS * HEADS * HPAD * 2);

  // weight prep (bf16 conversion / packing)
  {
    int n = LAYERS * DMODEL * 4 * DMODEL;
    pack_qkvg_kernel<<<(n + 255) / 256, 256, 0, stream>>>(Wq, Wk, Wv, Wg, wqkvg);
    n = LAYERS * DMODEL * DMODEL;
    f32_to_bf16_kernel<<<(n + 255) / 256, 256, 0, stream>>>(Wo, wob, n);
    n = LAYERS * DMODEL * FDIM;
    f32_to_bf16_kernel<<<(n + 255) / 256, 256, 0, stream>>>(W1, w1b, n);
    n = LAYERS * FDIM * DMODEL;
    f32_to_bf16_kernel<<<(n + 255) / 256, 256, 0, stream>>>(W2, w2b, n);
    n = DMODEL * OUTD;
    f32_to_bf16_kernel<<<(n + 255) / 256, 256, 0, stream>>>(Wenc, wencb, n);
  }

  hipMemcpyAsync(XF, x, (size_t)ROWS * DMODEL * 4, hipMemcpyDeviceToDevice, stream);

  const int nD = ROWS * DMODEL;
  const int nF = ROWS * FDIM;

  for (int l = 0; l < LAYERS; ++l) {
    // Xn = LN(X) -> bf16
    layernorm_bf16_kernel<<<ROWS / 8, 256, 0, stream>>>(XF, ln_g, ln_b, ABuf);
    // QKVG = Xn @ [Wq|Wk|Wv|Wg]
    gemm_bf16_kernel<<<dim3(4 * DMODEL / 64, ROWS / 64), 128, 0, stream>>>(
        ABuf, wqkvg + (size_t)l * DMODEL * 4 * DMODEL, CF, ROWS, 4 * DMODEL, DMODEL, nullptr);
    // pack heads (pad HD 24 -> 32, bf16; V transposed)
    {
      int n = ROWS * HEADS * HPAD;
      repack_qkv_kernel<<<(n + 255) / 256, 256, 0, stream>>>(CF, Qb, Kb, Vb);
    }
    // causal retention with decay (flash-style)
    retention_kernel<<<dim3(SEQ / 64, HEADS, BATCH), 128, 0, stream>>>(Qb, Kb, Vb, YR);
    // group norm + silu gate -> bf16
    {
      int n = ROWS * HEADS;
      gn_gate_kernel<<<(n + 255) / 256, 256, 0, stream>>>(
          YR, CF, gn_w + (size_t)l * DMODEL, gn_b + (size_t)l * DMODEL, ABuf);
    }
    // Y = gated @ Wo + X
    gemm_bf16_kernel<<<dim3(DMODEL / 64, ROWS / 64), 128, 0, stream>>>(
        ABuf, wob + (size_t)l * DMODEL * DMODEL, YF, ROWS, DMODEL, DMODEL, nullptr);
    add_inplace_kernel<<<(nD + 255) / 256, 256, 0, stream>>>(YF, XF, nD);
    // FFN: X = gelu(LN(Y) @ W1) @ W2 + Y
    layernorm_bf16_kernel<<<ROWS / 8, 256, 0, stream>>>(YF, ln_g, ln_b, ABuf);
    gemm_bf16_kernel<<<dim3(FDIM / 64, ROWS / 64), 128, 0, stream>>>(
        ABuf, w1b + (size_t)l * DMODEL * FDIM, CF, ROWS, FDIM, DMODEL, nullptr);
    gelu_bf16_kernel<<<(nF + 255) / 256, 256, 0, stream>>>(CF, ABuf, nF);
    gemm_bf16_kernel<<<dim3(DMODEL / 64, ROWS / 64), 128, 0, stream>>>(
        ABuf, w2b + (size_t)l * FDIM * DMODEL, XF, ROWS, DMODEL, FDIM, nullptr);
    add_inplace_kernel<<<(nD + 255) / 256, 256, 0, stream>>>(XF, YF, nD);
  }

  // out = X @ Wenc + benc
  f32_to_bf16_kernel<<<(nD + 255) / 256, 256, 0, stream>>>(XF, ABuf, nD);
  gemm_bf16_kernel<<<dim3(OUTD / 64, ROWS / 64), 128, 0, stream>>>(
      ABuf, wencb, (float*)d_out, ROWS, OUTD, DMODEL, benc);
}
